// GroupedVectorAttention_68985764708923
// MI455X (gfx1250) — compile-verified, matrix operand order checked by reading, not run on hardware
//
#include <hip/hip_runtime.h>
#include <hip/hip_bf16.h>
#include <math.h>

#define NPTS 100000
#define S 16
#define C 256
#define G 16
#define EPS 1e-5f
#define PTS 8

typedef __attribute__((ext_vector_type(16))) __bf16 v16bf;
typedef __attribute__((ext_vector_type(8)))  __bf16 v8bf;
typedef __attribute__((ext_vector_type(8)))  float  v8f;

union V16U { v16bf v; v8bf h[2]; };

__device__ __forceinline__ __bf16 f2bf(float f) {
  unsigned u = __builtin_bit_cast(unsigned, f);
  unsigned r = u + 0x7FFFu + ((u >> 16) & 1u);   // round-to-nearest-even
  unsigned short s = (unsigned short)(r >> 16);
  return __builtin_bit_cast(__bf16, s);
}

// Low 32 bits of a generic pointer to __shared__ = LDS byte offset (aperture mapping).
__device__ __forceinline__ unsigned ldsOff(const void* p) {
  return (unsigned)(unsigned long long)p;
}

// ASYNCcnt-tracked DMA: 16 bytes global -> LDS, no VGPR staging.
__device__ __forceinline__ void asyncLd16(unsigned lds_byte_off, const void* gaddr) {
  asm volatile("global_load_async_to_lds_b128 %0, %1, off"
               :: "v"(lds_byte_off), "v"(gaddr)
               : "memory");
}
__device__ __forceinline__ void waitAsync() {
  asm volatile("s_wait_asynccnt 0x0" ::: "memory");
}

// A fragment (16x32 bf16) from LDS. ISA 16-bit A layout: lane L holds row M=L%16,
// K chunks [c0..c0+7] and [c0+16..c0+23] with c0 = (L/16)*8.
__device__ __forceinline__ v16bf ldsA(const __bf16* base, int stride, int lane, int kbase) {
  int r  = lane & 15;
  int kk = (lane >> 4) << 3;
  const __bf16* p = base + r * stride + kbase + kk;
  V16U u;
  u.h[0] = *(const v8bf*)(p);
  u.h[1] = *(const v8bf*)(p + 16);
  return u.v;
}

// B fragment (32x16 bf16) from pre-transposed weights WT[n][k] (row-major, row=out-col).
__device__ __forceinline__ v16bf ldgB(const __bf16* WT, int ct, int lane, int kbase) {
  int cl = lane & 15;
  int kk = (lane >> 4) << 3;
  const __bf16* p = WT + (size_t)(ct * 16 + cl) * C + kbase + kk;
  V16U u;
  u.h[0] = *(const v8bf*)(p);
  u.h[1] = *(const v8bf*)(p + 16);
  return u.v;
}

// ---------------- K0: zero stats region (incl. zero pad row used as DMA source) ----------
__global__ void k_zero(float* stats) {
  int i = blockIdx.x * blockDim.x + threadIdx.x;
  if (i < 4096) stats[i] = 0.f;
}

// ---------------- K1a: transpose + convert weights to bf16 ----------------
__global__ void k_wt(const float* Wq, const float* Wk, const float* Wv, const float* Wp2,
                     __bf16* wt) {
  int n = blockIdx.x, k = threadIdx.x, m = blockIdx.y;
  const float* W = (m == 0) ? Wq : (m == 1) ? Wk : (m == 2) ? Wv : Wp2;
  __bf16* WT = wt + (size_t)m * C * C;
  WT[n * C + k] = f2bf(W[k * C + n]);
}

// ---------------- K1b: feat f32 -> bf16 (so GEMM A-tiles are pure DMA) ----------------
__global__ void k_feat2bf(const float* __restrict__ feat, __bf16* __restrict__ featbf) {
  size_t i = ((size_t)blockIdx.x * blockDim.x + threadIdx.x) * 4;
#pragma unroll
  for (int j = 0; j < 4; ++j) featbf[i + j] = f2bf(feat[i + j]);
}

// ---------------- K2: q/k/v = feat @ W + b  (WMMA bf16, async A-tile fill) ----------------
__global__ void k_gemm_qkv(const __bf16* __restrict__ featbf, const __bf16* __restrict__ wt,
                           const float* bq, const float* bk, const float* bv,
                           float* qlin, float* klin, float* vlin) {
  __shared__ __align__(16) __bf16 As[16][264];
  int m = blockIdx.y;
  const __bf16* WT = wt + (size_t)m * C * C;
  const float* bias = (m == 0) ? bq : (m == 1) ? bk : bv;
  float* out = (m == 0) ? qlin : (m == 1) ? klin : vlin;
  int rt = blockIdx.x, tid = threadIdx.x;

  __builtin_prefetch(WT + (size_t)tid * 256, 0, 0);          // warm B weights (L2->near)
  __builtin_prefetch(WT + (size_t)tid * 256 + 128, 0, 0);

  // Async DMA the 16x256 bf16 A tile: row r = tid>>4, two 16B chunks per thread.
  {
    int r = tid >> 4, ch = tid & 15;
    const __bf16* src = featbf + (size_t)(rt * 16 + r) * C;
    unsigned dst = ldsOff(&As[r][0]);
    asyncLd16(dst + ch * 16,       (const char*)src + ch * 16);
    asyncLd16(dst + ch * 16 + 256, (const char*)src + ch * 16 + 256);
  }
  waitAsync();
  __syncthreads();

  int wv = tid >> 5, lane = tid & 31;
#pragma unroll
  for (int t = 0; t < 2; ++t) {
    int ct = wv * 2 + t;
    v8f acc = {};
#pragma unroll
    for (int kb = 0; kb < 8; ++kb) {
      v16bf a = ldsA(&As[0][0], 264, lane, kb * 32);
      v16bf b = ldgB(WT, ct, lane, kb * 32);
      acc = __builtin_amdgcn_wmma_f32_16x16x32_bf16(false, a, false, b, (short)0, acc, false, false);
    }
    int c = ct * 16 + (lane & 15);
    float bc = bias[c];
#pragma unroll
    for (int rr = 0; rr < 8; ++rr) {
      int mm = rr + ((lane >> 4) << 3);
      out[(size_t)(rt * 16 + mm) * C + c] = acc[rr] + bc;
    }
  }
}

// ---------------- K3: per-channel sum/sumsq of q_lin, k_lin ----------------
__global__ void k_stats_qk(const float* __restrict__ qlin, const float* __restrict__ klin,
                           float* stats) {
  int mtx = blockIdx.y;
  const float* p = (mtx == 0) ? qlin : klin;
  int c = threadIdx.x;
  int r0 = blockIdx.x * 512;
  float s1 = 0.f, s2 = 0.f;
  for (int r = r0; r < r0 + 512 && r < NPTS; ++r) {
    float x = p[(size_t)r * C + c];
    s1 += x; s2 += x * x;
  }
  atomicAdd(&stats[mtx * 512 + c], s1);
  atomicAdd(&stats[mtx * 512 + 256 + c], s2);
}

// ---------------- K4: global moments of pos (3-vec sum + 3x3 second moment) ----------------
__global__ void k_pos_stats(const float* __restrict__ coord, const int* __restrict__ ridx,
                            float* stats) {
  long gid = (long)blockIdx.x * blockDim.x + threadIdx.x;
  long stride = (long)gridDim.x * blockDim.x;
  float a0 = 0, a1 = 0, a2 = 0, m00 = 0, m01 = 0, m02 = 0, m11 = 0, m12 = 0, m22 = 0;
  for (long p = gid; p < (long)NPTS * S; p += stride) {
    int n = (int)(p >> 4);
    int idx = ridx[p];
    if (idx >= 0) {   // idx == -1 -> pos is 0 (masked), contributes nothing
      float dx = coord[idx * 3 + 0] - coord[n * 3 + 0];
      float dy = coord[idx * 3 + 1] - coord[n * 3 + 1];
      float dz = coord[idx * 3 + 2] - coord[n * 3 + 2];
      a0 += dx; a1 += dy; a2 += dz;
      m00 += dx * dx; m01 += dx * dy; m02 += dx * dz;
      m11 += dy * dy; m12 += dy * dz; m22 += dz * dz;
    }
  }
  for (int o = 16; o > 0; o >>= 1) {
    a0 += __shfl_xor(a0, o);  a1 += __shfl_xor(a1, o);  a2 += __shfl_xor(a2, o);
    m00 += __shfl_xor(m00, o); m01 += __shfl_xor(m01, o); m02 += __shfl_xor(m02, o);
    m11 += __shfl_xor(m11, o); m12 += __shfl_xor(m12, o); m22 += __shfl_xor(m22, o);
  }
  if ((threadIdx.x & 31) == 0) {
    atomicAdd(&stats[1024], a0);  atomicAdd(&stats[1025], a1);  atomicAdd(&stats[1026], a2);
    atomicAdd(&stats[1027], m00); atomicAdd(&stats[1028], m01); atomicAdd(&stats[1029], m02);
    atomicAdd(&stats[1030], m11); atomicAdd(&stats[1031], m12); atomicAdd(&stats[1032], m22);
  }
}

// ---------------- K5: fold BN params (q, k, pos-hidden) ----------------
__global__ void k_finalize(const float* gq, const float* betaq,
                           const float* gk, const float* betak,
                           const float* Wp1, const float* bp1,
                           const float* gp, const float* betap,
                           float* stats) {
  int c = threadIdx.x;
  float invN = 1.f / (float)NPTS;
  float mq = stats[c] * invN;
  float vq = stats[256 + c] * invN - mq * mq;
  float aqv = gq[c] * rsqrtf(vq + EPS);
  stats[1152 + c] = aqv;
  stats[1408 + c] = betaq[c] - mq * aqv;
  float mk = stats[512 + c] * invN;
  float vk = stats[768 + c] * invN - mk * mk;
  float akv = gk[c] * rsqrtf(vk + EPS);
  stats[1664 + c] = akv;
  stats[1920 + c] = betak[c] - mk * akv;
  // hidden = relu(bn(pos @ Wp1 + bp1)); stats from pos moments (pos is 3-dim)
  float w0 = Wp1[c], w1 = Wp1[C + c], w2 = Wp1[2 * C + c];
  float P0 = stats[1024], P1 = stats[1025], P2 = stats[1026];
  float xx = stats[1027], xy = stats[1028], xz = stats[1029];
  float yy = stats[1030], yz = stats[1031], zz = stats[1032];
  float cnt = (float)NPTS * (float)S;
  float b = bp1[c];
  float wP = w0 * P0 + w1 * P1 + w2 * P2;
  float mean = wP / cnt + b;
  float wMw = w0 * w0 * xx + w1 * w1 * yy + w2 * w2 * zz +
              2.f * (w0 * w1 * xy + w0 * w2 * xz + w1 * w2 * yz);
  float ex2 = (wMw + 2.f * b * wP) / cnt + b * b;
  float var = ex2 - mean * mean;
  float sc = gp[c] * rsqrtf(var + EPS);
  float4 hc;
  hc.x = w0 * sc; hc.y = w1 * sc; hc.z = w2 * sc;
  hc.w = b * sc + betap[c] - mean * sc;
  ((float4*)(stats + 2176))[c] = hc;
}

// ---------------- K6: fused w_raw = grouped( (k_g - q + peb) * w_gl ) + group stats --------
// k rows DMA'd raw into LDS (idx<0 -> zero row); BN+relu+mask applied at the WMMA epilogue.
__global__ void k_wraw(const float* __restrict__ coord, const int* __restrict__ ridx,
                       const float* __restrict__ klin, const float* __restrict__ qlin,
                       const __bf16* __restrict__ Wp2T, const float* bp2, const float* wgl,
                       const float* __restrict__ zrow,
                       float* stats, float* wraw) {
  __shared__ __align__(16) __bf16 Hs[16][264];
  __shared__ __align__(16) float Kg[16][260];
  __shared__ float qv[256];
  __shared__ float msk[16];
  __shared__ float gs1[16], gs2[16];
  int tid = threadIdx.x;
  if (tid < 16) { gs1[tid] = 0.f; gs2[tid] = 0.f; }
  const float4* hc = (const float4*)(stats + 2176);
  const float* aq = stats + 1152; const float* bqs = stats + 1408;
  const float* ak = stats + 1664; const float* bks = stats + 1920;
  int s = tid >> 4, lq = tid & 15, c0 = lq << 4;
  int wv = tid >> 5, lane = tid & 31;

  __builtin_prefetch(Wp2T + (size_t)tid * 256, 0, 0);
  __builtin_prefetch(Wp2T + (size_t)tid * 256 + 128, 0, 0);

  for (int p = 0; p < PTS; ++p) {
    int n = blockIdx.x * PTS + p;
    __syncthreads();
    {
      int idx = ridx[n * S + s];
      // raw k row (or zero pad row) -> LDS via async DMA: 4x16B per thread
      {
        const float* src = (idx >= 0) ? (klin + (size_t)idx * C) : zrow;
        unsigned dst = ldsOff(&Kg[s][0]);
#pragma unroll
        for (int j = 0; j < 4; ++j) {
          int bo = (lq + 16 * j) * 16;
          asyncLd16(dst + bo, (const char*)src + bo);
        }
      }
      float px = 0.f, py = 0.f, pz = 0.f;
      if (idx >= 0) {
        px = coord[idx * 3 + 0] - coord[n * 3 + 0];
        py = coord[idx * 3 + 1] - coord[n * 3 + 1];
        pz = coord[idx * 3 + 2] - coord[n * 3 + 2];
      }
#pragma unroll 4
      for (int cc = 0; cc < 16; ++cc) {
        int c = c0 + cc;
        float4 h = hc[c];
        Hs[s][c] = f2bf(fmaxf(h.x * px + h.y * py + h.z * pz + h.w, 0.f));
      }
      if (lq == 0) msk[s] = (idx >= 0) ? 1.f : 0.f;
      if (s == 0) {
#pragma unroll 4
        for (int cc = 0; cc < 16; ++cc) {
          int c = c0 + cc;
          qv[c] = fmaxf(qlin[(size_t)n * C + c] * aq[c] + bqs[c], 0.f);
        }
      }
    }
    waitAsync();
    __syncthreads();
#pragma unroll
    for (int t = 0; t < 2; ++t) {
      int g = wv * 2 + t;     // channel group == column tile
      v8f acc = {};
#pragma unroll
      for (int kb = 0; kb < 8; ++kb) {
        v16bf a = ldsA(&Hs[0][0], 264, lane, kb * 32);
        v16bf b = ldgB(Wp2T, g, lane, kb * 32);
        acc = __builtin_amdgcn_wmma_f32_16x16x32_bf16(false, a, false, b, (short)0, acc, false, false);
      }
      int c = g * 16 + (lane & 15);
      float bp2c = bp2[c], wglc = wgl[c], qc = qv[c];
      float akc = ak[c], bkc = bks[c];
#pragma unroll
      for (int rr = 0; rr < 8; ++rr) {
        int sr = rr + ((lane >> 4) << 3);
        float kval = msk[sr] * fmaxf(Kg[sr][c] * akc + bkc, 0.f);  // bn+relu+pad at use
        float val = (kval - qc + acc[rr] + bp2c) * wglc;
        val += __shfl_xor(val, 1);
        val += __shfl_xor(val, 2);
        val += __shfl_xor(val, 4);
        val += __shfl_xor(val, 8);
        if ((lane & 15) == 0) {
          wraw[((size_t)n * S + sr) * G + g] = val;
          atomicAdd(&gs1[g], val);
          atomicAdd(&gs2[g], val * val);
        }
      }
    }
  }
  __syncthreads();
  if (tid < 16) {
    atomicAdd(&stats[1040 + tid], gs1[tid]);
    atomicAdd(&stats[1056 + tid], gs2[tid]);
  }
}

// ---------------- K7: fold group BN ----------------
__global__ void k_gbn(const float* g_gl, const float* beta_gl, float* stats) {
  int g = threadIdx.x;
  if (g < 16) {
    float cnt = (float)NPTS * (float)S;
    float mean = stats[1040 + g] / cnt;
    float var = stats[1056 + g] / cnt - mean * mean;
    float sc = g_gl[g] * rsqrtf(var + EPS);
    stats[1088 + g] = sc;
    stats[1104 + g] = beta_gl[g] - mean * sc;
  }
}

// ---------------- K8: fused Wwe + masked softmax + output accumulation ----------------
__global__ void k_out(const float* __restrict__ coord, const int* __restrict__ ridx,
                      const float* __restrict__ vlin, const float* __restrict__ wraw,
                      const __bf16* __restrict__ Wp2T, const float* bp2,
                      const float* Wwe, const float* bwe,
                      const float* __restrict__ zrow,
                      const float* stats, float* out) {
  __shared__ __align__(16) __bf16 Hs[16][264];
  __shared__ __align__(16) float Vg[16][260];
  __shared__ float tb[16][16];
  __shared__ float sm[16][17];
  __shared__ float msk[16];
  int tid = threadIdx.x;
  const float4* hc = (const float4*)(stats + 2176);
  const float* gsc = stats + 1088; const float* gsh = stats + 1104;
  int s = tid >> 4, lq = tid & 15, c0 = lq << 4;
  int wv = tid >> 5, lane = tid & 31;

  __builtin_prefetch(Wp2T + (size_t)tid * 256, 0, 0);
  __builtin_prefetch(Wp2T + (size_t)tid * 256 + 128, 0, 0);

  for (int p = 0; p < PTS; ++p) {
    int n = blockIdx.x * PTS + p;
    __syncthreads();
    {
      int idx = ridx[n * S + s];
      // raw v row (pad row is exactly zeros, per reference) -> LDS via async DMA
      {
        const float* src = (idx >= 0) ? (vlin + (size_t)idx * C) : zrow;
        unsigned dst = ldsOff(&Vg[s][0]);
#pragma unroll
        for (int j = 0; j < 4; ++j) {
          int bo = (lq + 16 * j) * 16;
          asyncLd16(dst + bo, (const char*)src + bo);
        }
      }
      float px = 0.f, py = 0.f, pz = 0.f;
      if (idx >= 0) {
        px = coord[idx * 3 + 0] - coord[n * 3 + 0];
        py = coord[idx * 3 + 1] - coord[n * 3 + 1];
        pz = coord[idx * 3 + 2] - coord[n * 3 + 2];
      }
#pragma unroll 4
      for (int cc = 0; cc < 16; ++cc) {
        int c = c0 + cc;
        float4 h = hc[c];
        Hs[s][c] = f2bf(fmaxf(h.x * px + h.y * py + h.z * pz + h.w, 0.f));
      }
      if (lq == 0) msk[s] = (idx >= 0) ? 1.f : 0.f;
      tb[s][lq] = fmaxf(wraw[((size_t)n * S + s) * G + lq] * gsc[lq] + gsh[lq], 0.f);
    }
    waitAsync();
    __syncthreads();
    {
      float acc2 = bwe[lq];
#pragma unroll
      for (int g = 0; g < 16; ++g) acc2 += tb[s][g] * Wwe[g * G + lq];
      sm[s][lq] = acc2;
    }
    __syncthreads();
    if (tid < 16) {  // softmax over S per group column (mask applied AFTER, per reference)
      float mx = -1e30f;
      for (int ss = 0; ss < 16; ++ss) mx = fmaxf(mx, sm[ss][tid]);
      float sum = 0.f;
      for (int ss = 0; ss < 16; ++ss) { float e = __expf(sm[ss][tid] - mx); sm[ss][tid] = e; sum += e; }
      float inv = 1.f / sum;
      for (int ss = 0; ss < 16; ++ss) sm[ss][tid] *= inv * msk[ss];
    }
    __syncthreads();
#pragma unroll
    for (int t = 0; t < 2; ++t) {
      int g = wv * 2 + t;
      v8f acc = {};
#pragma unroll
      for (int kb = 0; kb < 8; ++kb) {
        v16bf a = ldsA(&Hs[0][0], 264, lane, kb * 32);
        v16bf b = ldgB(Wp2T, g, lane, kb * 32);
        acc = __builtin_amdgcn_wmma_f32_16x16x32_bf16(false, a, false, b, (short)0, acc, false, false);
      }
      int c = g * 16 + (lane & 15);
      float bp2c = bp2[c];
      float partial = 0.f;
#pragma unroll
      for (int rr = 0; rr < 8; ++rr) {
        int sr = rr + ((lane >> 4) << 3);
        partial += (Vg[sr][c] + acc[rr] + bp2c) * sm[sr][g];
      }
      partial += __shfl_xor(partial, 16);
      if (lane < 16) out[(size_t)n * C + c] = partial;
    }
  }
}

extern "C" void kernel_launch(void* const* d_in, const int* in_sizes, int n_in,
                              void* d_out, int out_size, void* d_ws, size_t ws_size,
                              hipStream_t stream) {
  (void)in_sizes; (void)n_in; (void)out_size; (void)ws_size;
  const float* feat   = (const float*)d_in[0];
  const float* coord  = (const float*)d_in[1];
  const int*   ridx   = (const int*)d_in[2];
  const float* Wq = (const float*)d_in[3];  const float* bq = (const float*)d_in[4];
  const float* gq = (const float*)d_in[5];  const float* betaq = (const float*)d_in[6];
  const float* Wk = (const float*)d_in[7];  const float* bk = (const float*)d_in[8];
  const float* gk = (const float*)d_in[9];  const float* betak = (const float*)d_in[10];
  const float* Wv = (const float*)d_in[11]; const float* bv = (const float*)d_in[12];
  const float* Wp1 = (const float*)d_in[13]; const float* bp1 = (const float*)d_in[14];
  const float* gp = (const float*)d_in[15];  const float* betap = (const float*)d_in[16];
  const float* Wp2 = (const float*)d_in[17]; const float* bp2 = (const float*)d_in[18];
  const float* wgl = (const float*)d_in[19];
  const float* g_gl = (const float*)d_in[20]; const float* beta_gl = (const float*)d_in[21];
  const float* Wwe = (const float*)d_in[22];  const float* bwe = (const float*)d_in[23];

  char* ws = (char*)d_ws;
  __bf16* wt   = (__bf16*)ws;                       // 4 x 256x256 bf16 = 512 KB
  float* stats = (float*)(ws + 524288);             // 16 KB stats/folded-BN region
  const size_t MAT = (size_t)NPTS * C * sizeof(float); // 102.4 MB
  float* qlin = (float*)(ws + 540672);
  float* klin = (float*)(ws + 540672 + MAT);
  float* vlin = (float*)(ws + 540672 + 2 * MAT);
  float* wraw = (float*)(ws + 540672 + 3 * MAT);    // N*S*G f32 (same size as MAT)
  __bf16* featbf = (__bf16*)(ws + 540672 + 4 * MAT); // N*C bf16 = 51.2 MB
  const __bf16* Wp2T = wt + 3 * (size_t)C * C;
  const float* zrow = stats + 3200;                  // 256 zeros (k_zero clears 4096 floats)

  k_zero<<<16, 256, 0, stream>>>(stats);
  k_wt<<<dim3(256, 4), 256, 0, stream>>>(Wq, Wk, Wv, Wp2, wt);
  k_feat2bf<<<(NPTS * C) / 1024, 256, 0, stream>>>(feat, featbf);
  k_gemm_qkv<<<dim3(NPTS / 16, 3), 256, 0, stream>>>(featbf, wt, bq, bk, bv, qlin, klin, vlin);
  k_stats_qk<<<dim3((NPTS + 511) / 512, 2), 256, 0, stream>>>(qlin, klin, stats);
  k_pos_stats<<<1024, 256, 0, stream>>>(coord, ridx, stats);
  k_finalize<<<1, 256, 0, stream>>>(gq, betaq, gk, betak, Wp1, bp1, gp, betap, stats);
  k_wraw<<<NPTS / PTS, 256, 0, stream>>>(coord, ridx, klin, qlin, Wp2T, bp2, wgl, zrow,
                                         stats, wraw);
  k_gbn<<<1, 32, 0, stream>>>(g_gl, beta_gl, stats);
  k_out<<<NPTS / PTS, 256, 0, stream>>>(coord, ridx, vlin, wraw, Wp2T, bp2, Wwe, bwe, zrow,
                                        stats, (float*)d_out);
}
